// ContrastiveLoss_44530220925474
// MI455X (gfx1250) — matrix operand order repaired and checked by reading, not compiled, not run on hardware
//
#include <hip/hip_runtime.h>

// ---------------------------------------------------------------------------
// NT-Xent contrastive loss, fused flash-style on gfx1250 (MI455X, wave32).
//   B = 4096, D = 256, T = 0.5  ->  2B = 8192 rows; sim (8192x8192) is never
//   materialized.  Main GEMM: v_wmma_f32_16x16x32_f16 (two independent tile
//   chains per iteration), B tiles double-buffered in LDS via async
//   global->LDS (ASYNCcnt), raw v_exp_f32 for exp.
// ---------------------------------------------------------------------------

#define CB   4096      // batch
#define CD   256       // embedding dim
#define NR   8192      // 2B rows
#define CPI  32        // columns per iteration (two 16-col WMMA tiles)
#define NI   (NR / CPI)            // 256 iterations
#define LDSTRIDE 264   // 256 + 8 halves pad: 528 B row stride => 16-byte
                       // fragment reads from 16 lanes sweep all 64 LDS banks
#define INV_T_LOG2E 2.8853900817779268f   // (1/T) * log2(e) = 2*log2(e)

// -- gfx1250 async global->LDS staging (guarded; falls back to VGPR staging) --
#ifndef NTX_ASYNC
#if defined(__HIP_DEVICE_COMPILE__) && defined(__gfx1250__) && \
    __has_builtin(__builtin_amdgcn_global_load_async_to_lds_b128)
#define NTX_ASYNC 1
#else
#define NTX_ASYNC 0
#endif
#endif

typedef __attribute__((ext_vector_type(16))) _Float16 v16h;
typedef __attribute__((ext_vector_type(8)))  _Float16 v8h;
typedef __attribute__((ext_vector_type(8)))  float    v8f;

// exact parameter types of __builtin_amdgcn_global_load_async_to_lds_b128
typedef int v4i_vs __attribute__((vector_size(16)));
typedef __attribute__((address_space(1))) v4i_vs* async_gptr;
typedef __attribute__((address_space(3))) v4i_vs* async_lptr;

static __device__ __forceinline__ v16h cat16(v8h lo, v8h hi) {
    return __builtin_shufflevector(lo, hi, 0,1,2,3,4,5,6,7,8,9,10,11,12,13,14,15);
}

static __device__ __forceinline__ float fast_exp2(float x) {
#if defined(__HIP_DEVICE_COMPILE__) && __has_builtin(__builtin_amdgcn_exp2f)
    return __builtin_amdgcn_exp2f(x);   // bare v_exp_f32: |x| <= 2.89, no fixup needed
#else
    return exp2f(x);
#endif
}

static __device__ __forceinline__ void wait_stage() {
#if NTX_ASYNC
#if __has_builtin(__builtin_amdgcn_s_wait_asynccnt)
    __builtin_amdgcn_s_wait_asynccnt(0);
#else
    asm volatile("s_wait_asynccnt 0x0" ::: "memory");
#endif
#endif
}

// Stage one 32-col x 256-K f16 tile (16 KB) into LDS; 128 threads x 8 chunks.
static __device__ __forceinline__ void stage_tile32(const _Float16* __restrict__ Z,
                                                    _Float16* dst, int t, int it) {
    #pragma unroll
    for (int i = 0; i < 8; ++i) {
        const int idx  = t + i * 128;        // 0..1023
        const int c    = idx >> 5;           // column within tile (0..31)
        const int koff = (idx & 31) * 8;     // K offset (16 B chunks)
#if NTX_ASYNC
        __builtin_amdgcn_global_load_async_to_lds_b128(
            (async_gptr)(Z + (size_t)(it * CPI + c) * CD + koff),
            (async_lptr)(dst + c * LDSTRIDE + koff),
            0, 0);
#else
        *(v8h*)(dst + c * LDSTRIDE + koff) =
            *(const v8h*)(Z + (size_t)(it * CPI + c) * CD + koff);
#endif
    }
}

static __device__ __forceinline__ float block_reduce_sum(float v, float* red) {
    const int t = threadIdx.x;
    red[t] = v;
    __syncthreads();
    #pragma unroll
    for (int s = 128; s > 0; s >>= 1) {
        if (t < s) red[t] += red[t + s];
        __syncthreads();
    }
    float r = red[0];
    __syncthreads();
    return r;
}

// ---------------------------------------------------------------------------
// Kernel 1: L2-normalize emb_i / emb_j, emit Z in f16 (row-major [8192][256]),
// and exact fp32 positives pos[n] = dot(z_i[n], z_j[n]).
// ---------------------------------------------------------------------------
__global__ void __launch_bounds__(256) ntxent_normalize(
        const float* __restrict__ emb_i, const float* __restrict__ emb_j,
        _Float16* __restrict__ Z, float* __restrict__ pos) {
    __shared__ float red[256];
    const int n = blockIdx.x;
    const int t = threadIdx.x;

    const float xi = emb_i[(size_t)n * CD + t];
    const float xj = emb_j[(size_t)n * CD + t];

    const float si = block_reduce_sum(xi * xi, red);
    const float sj = block_reduce_sum(xj * xj, red);

    const float zi = xi / fmaxf(sqrtf(si), 1e-12f);   // matches F.normalize
    const float zj = xj / fmaxf(sqrtf(sj), 1e-12f);

    Z[(size_t)n        * CD + t] = (_Float16)zi;
    Z[(size_t)(n + CB) * CD + t] = (_Float16)zj;

    const float p = block_reduce_sum(zi * zj, red);
    if (t == 0) pos[n] = p;
}

// ---------------------------------------------------------------------------
// Kernel 2: fused  rowsum[m] = sum_{n != m} exp(2 * <Z[m], Z[n]>)
// Block = 4 waves (128 threads) = 64 rows.  Wave w owns 16 rows; its A
// fragments (16x256 f16, pre-scaled by 2*log2e) live in registers for the
// whole column sweep.  Each iteration consumes a 32-column LDS tile as two
// independent 16x16x(K=256) WMMA chains while the next tile is staged
// asynchronously into the other buffer.  No atomics -> deterministic.
// ---------------------------------------------------------------------------
__global__ void __launch_bounds__(128) ntxent_simrows(
        const _Float16* __restrict__ Z, float* __restrict__ rowsum) {
    __shared__ __align__(16) _Float16 btile[2][CPI * LDSTRIDE];

    const int t    = threadIdx.x;
    const int lane = t & 31;
    const int w    = t >> 5;
    const int l16  = lane & 15;
    const int half = lane >> 4;          // 0: lanes 0-15, 1: lanes 16-31

    const int rowtile = blockIdx.x * 4 + w;
    const int rowbase = rowtile * 16;

    // ---- preload A fragments (wave32 16x32 f16 A layout), scale by 1/T*log2e
    // lane L: M = L%16;  e=0..7  -> K = kc*32 + (L/16)*8 + e
    //                    e=8..15 -> K = kc*32 + 16 + (L/16)*8 + (e-8)
    const _Float16* arow = Z + (size_t)(rowbase + l16) * CD;
    const _Float16 hscale = (_Float16)INV_T_LOG2E;
    v16h a[8];
    #pragma unroll
    for (int kc = 0; kc < 8; ++kc) {
        const int k0 = kc * 32 + half * 8;
        v8h lo = *(const v8h*)(arow + k0);
        v8h hi = *(const v8h*)(arow + k0 + 16);
        a[kc] = cat16(lo, hi) * hscale;
    }

    float acc[8];
    #pragma unroll
    for (int g = 0; g < 8; ++g) acc[g] = 0.0f;

    // ---- software pipeline: stage tile 0, then overlap stage(it+1)/compute(it)
    stage_tile32(Z, &btile[0][0], t, 0);
    wait_stage();
    __syncthreads();

    for (int it = 0; it < NI; ++it) {
        const int cur = it & 1;
        if (it + 1 < NI) stage_tile32(Z, &btile[cur ^ 1][0], t, it + 1);

        // ---- two independent 16x16 accumulator chains over K = 256 ----
        // B 32x16 f16 layout: lane L -> N = L%16, e<8: K = kc*32 + (L/16)*8 + e, e>=8: K = kc*32 + 16 + (L/16)*8 + (e-8)
        const _Float16* bb0 = &btile[cur][l16 * LDSTRIDE + half * 8];
        const _Float16* bb1 = bb0 + 16 * LDSTRIDE;
        v8f c0 = {}, c1 = {};
        #pragma unroll
        for (int kc = 0; kc < 8; ++kc) {
            v16h b0 = cat16(*(const v8h*)(bb0 + kc * 32),
                            *(const v8h*)(bb0 + kc * 32 + 16));
            v16h b1 = cat16(*(const v8h*)(bb1 + kc * 32),
                            *(const v8h*)(bb1 + kc * 32 + 16));
            c0 = __builtin_amdgcn_wmma_f32_16x16x32_f16(
                false, a[kc], false, b0, (short)0, c0, false, false);
            c1 = __builtin_amdgcn_wmma_f32_16x16x32_f16(
                false, a[kc], false, b1, (short)0, c1, false, false);
        }

        // ---- exp + accumulate; diagonal only in it == rowtile/2 ----
        if (it == (rowtile >> 1)) {
            const int sub = rowtile & 1;      // which 16-col subtile has the diag
            #pragma unroll
            for (int g = 0; g < 8; ++g) {
                const int  rloc = half * 8 + g;
                const float e0 = fast_exp2(c0[g]);   // scale folded into A
                const float e1 = fast_exp2(c1[g]);
                acc[g] += (sub == 0 && rloc == l16) ? 0.0f : e0;
                acc[g] += (sub == 1 && rloc == l16) ? 0.0f : e1;
            }
        } else {
            #pragma unroll
            for (int g = 0; g < 8; ++g)
                acc[g] += fast_exp2(c0[g]) + fast_exp2(c1[g]);
        }

        wait_stage();        // async writes to btile[cur^1] complete
        __syncthreads();     // everyone done reading cur; next buffer visible
    }

    // ---- deterministic cross-lane row reduction (16 N-lanes per half) ----
    #pragma unroll
    for (int g = 0; g < 8; ++g) {
        float v = acc[g];
        v += __shfl_xor(v, 1, 32);
        v += __shfl_xor(v, 2, 32);
        v += __shfl_xor(v, 4, 32);
        v += __shfl_xor(v, 8, 32);
        acc[g] = v;
    }
    if (l16 == 0) {
        #pragma unroll
        for (int g = 0; g < 8; ++g)
            rowsum[rowbase + half * 8 + g] = acc[g];
    }
}

// ---------------------------------------------------------------------------
// Kernel 3: loss = mean_n( log(denom_n) - pos_n / T ),  pos duplicated.
// ---------------------------------------------------------------------------
__global__ void __launch_bounds__(256) ntxent_finalize(
        const float* __restrict__ rowsum, const float* __restrict__ pos,
        float* __restrict__ out) {
    __shared__ float red[256];
    const int t = threadIdx.x;
    float s = 0.0f;
    for (int n = t; n < NR; n += 256) {
        s += logf(rowsum[n]) - 2.0f * pos[n & (CB - 1)];
    }
    red[t] = s;
    __syncthreads();
    #pragma unroll
    for (int sh = 128; sh > 0; sh >>= 1) {
        if (t < sh) red[t] += red[t + sh];
        __syncthreads();
    }
    if (t == 0) out[0] = red[0] * (1.0f / (float)NR);
}

// ---------------------------------------------------------------------------
extern "C" void kernel_launch(void* const* d_in, const int* in_sizes, int n_in,
                              void* d_out, int out_size, void* d_ws, size_t ws_size,
                              hipStream_t stream) {
    (void)in_sizes; (void)n_in; (void)out_size; (void)ws_size;

    const float* emb_i = (const float*)d_in[0];
    const float* emb_j = (const float*)d_in[1];
    float* out = (float*)d_out;

    // workspace: [ Z f16 (4 MB) | pos f32 (16 KB) | rowsum f32 (32 KB) ]
    _Float16* Z   = (_Float16*)d_ws;
    float* pos    = (float*)((char*)d_ws + (size_t)NR * CD * sizeof(_Float16));
    float* rowsum = pos + CB;

    ntxent_normalize<<<CB, 256, 0, stream>>>(emb_i, emb_j, Z, pos);
    ntxent_simrows<<<NR / 64, 128, 0, stream>>>(Z, rowsum);
    ntxent_finalize<<<1, 256, 0, stream>>>(rowsum, pos, out);
}
